// ResidualIntegrationNetworkRK4_39462159515992
// MI455X (gfx1250) — matrix operand order repaired and checked by reading, not compiled, not run on hardware
//
#include <hip/hip_runtime.h>

// ---------------------------------------------------------------------------
// ResidualIntegrationNetworkRK4 for MI455X (gfx1250, wave32, WMMA)
//
// Latency-bound serial scan: parallelism = BATCH(1024)/16 = 64 waves, one
// wave per workgroup, each owning 16 batch rows for all 16384 steps.
//  - Layer 1 (K=2): VALU f32 FMA + v_tanh_f32, written directly into the
//    16-bit A-fragment layout (ISA 7.12.2, 16-bit A 16x32 table).
//  - Layer 2 (64x64): v_wmma_f32_16x16x32_f16, 4 N-tiles x 2 K-chunks = 8
//    WMMAs per derivative eval, f32 accumulate.
//  - Layer 3 (N=1): fused tanh+W3 epilogue, 16-lane shfl_xor reduce, then a
//    register-only redistribution (cndmask chain + shfl_xor(16)).
//  - x[t+1] is software-prefetched one step ahead so the HBM load latency is
//    hidden behind a full step of compute (the scan state never waits on it).
// No LDS, no barriers anywhere in the scan loop; weights live in VGPRs.
// ---------------------------------------------------------------------------

typedef __attribute__((ext_vector_type(16))) _Float16 v16h;
typedef __attribute__((ext_vector_type(8)))  float    v8f;

#define SEQ_LEN 16384
#define BATCH   1024
#define HIDDEN  64

static __device__ __forceinline__ float fast_tanh(float x) {
#if __has_builtin(__builtin_amdgcn_tanhf)
  return __builtin_amdgcn_tanhf(x);
#elif __has_builtin(__builtin_amdgcn_tanh_f32)
  return __builtin_amdgcn_tanh_f32(x);
#else
  float r;
  asm volatile("v_tanh_f32 %0, %1" : "=v"(r) : "v"(x));
  return r;
#endif
}

__global__ __launch_bounds__(32) void rk4_scan_kernel(
    const float* __restrict__ x,   // [S, B, 1]
    const float* __restrict__ W1,  // [2, 64]
    const float* __restrict__ b1,  // [64]
    const float* __restrict__ W2,  // [64, 64]
    const float* __restrict__ b2,  // [64]
    const float* __restrict__ W3,  // [64, 1]
    const float* __restrict__ b3,  // [1]
    float* __restrict__ out)       // [S, B, 1]
{
  const int lane   = threadIdx.x;     // 0..31
  const int half   = lane >> 4;       // 0: lanes 0-15, 1: lanes 16-31
  const int lane16 = lane & 15;
  const int b0     = blockIdx.x * 16; // 16 batch rows per wave
  const int row    = b0 + lane16;

  // ---- Per-lane W1/b1 constants matching the f16 A-fragment K mapping ----
  // A 16x32 (16-bit): lanes 0-15: vgpr j<4 -> K=2j,2j+1 ; j>=4 -> K=2j+8,2j+9
  //                   lanes 16-31: same +8. Fragment c adds 32c.
  float w10[32], w11[32], b1c[32];
#pragma unroll
  for (int i = 0; i < 16; ++i) {
    const int c  = i >> 3;
    const int j  = i & 7;
    const int kb = 32 * c + ((j < 4) ? (2 * j) : (2 * j + 8)) + 8 * half;
    w10[2*i]   = W1[kb];          w10[2*i+1] = W1[kb + 1];
    w11[2*i]   = W1[64 + kb];     w11[2*i+1] = W1[64 + kb + 1];
    b1c[2*i]   = b1[kb];          b1c[2*i+1] = b1[kb + 1];
  }

  // ---- W2 B-fragments (f16), kept in VGPRs for the whole scan ----
  // B 32x16 (16-bit): lane -> N=lane16; lanes 0-15 hold K=0..7,16..23 (packed
  // pairs per VGPR), lanes 16-31 hold K=8..15,24..31. Chunk c adds 32c.
  v16h Bf[4][2];
#pragma unroll
  for (int nb = 0; nb < 4; ++nb) {
    const int col = 16 * nb + lane16;
#pragma unroll
    for (int c = 0; c < 2; ++c) {
#pragma unroll
      for (int e = 0; e < 16; ++e) {
        const int k = 32 * c + 8 * half + (e & 7) + 16 * (e >> 3);
        Bf[nb][c][e] = (_Float16)W2[k * HIDDEN + col];
      }
    }
  }

  // ---- layer-3 / bias constants ----
  float w3v[4], b2v[4];
#pragma unroll
  for (int nb = 0; nb < 4; ++nb) {
    w3v[nb] = W3[16 * nb + lane16];
    b2v[nb] = b2[16 * nb + lane16];
  }
  const float b3s = b3[0];

  // Redistribution predicates (loop-invariant).
  const int  sel = lane16 & 7;
  const bool midlane = (lane16 >= 8) != (half == 1);  // lanes 8..23

  float y = 0.0f;
  float vin = x[row];  // x for t=0 (FEAT==1, replicated in both halves)

  for (int t = 0; t < SEQ_LEN; ++t) {
    // Prefetch next step's x one full step ahead (independent of the scan
    // state) so the global-load latency is hidden behind this step's work.
    const int   tn       = (t + 1 < SEQ_LEN) ? (t + 1) : t;
    const float vin_next = x[tn * BATCH + row];

    float kv[4];
    float ya = y;  // second input of the derivative net

#pragma unroll
    for (int s = 0; s < 4; ++s) {
      // ---------- Layer 1: h1 = tanh([vin, ya] @ W1 + b1), in A layout ----
      v16h A0, A1;
#pragma unroll
      for (int i = 0; i < 8; ++i) {
        const float p0 = fmaf(vin, w10[2*i],   fmaf(ya, w11[2*i],   b1c[2*i]));
        const float p1 = fmaf(vin, w10[2*i+1], fmaf(ya, w11[2*i+1], b1c[2*i+1]));
        A0[2*i]   = (_Float16)fast_tanh(p0);
        A0[2*i+1] = (_Float16)fast_tanh(p1);
      }
#pragma unroll
      for (int i = 8; i < 16; ++i) {
        const int j = i - 8;
        const float p0 = fmaf(vin, w10[2*i],   fmaf(ya, w11[2*i],   b1c[2*i]));
        const float p1 = fmaf(vin, w10[2*i+1], fmaf(ya, w11[2*i+1], b1c[2*i+1]));
        A1[2*j]   = (_Float16)fast_tanh(p0);
        A1[2*j+1] = (_Float16)fast_tanh(p1);
      }

      // ---------- Layer 2 (WMMA) + fused layer-3 epilogue ----------------
      float acc[8] = {0.f, 0.f, 0.f, 0.f, 0.f, 0.f, 0.f, 0.f};
#pragma unroll
      for (int nb = 0; nb < 4; ++nb) {
        v8f cacc = {};
        cacc = __builtin_amdgcn_wmma_f32_16x16x32_f16(
            false, A0, false, Bf[nb][0], (short)0, cacc, false, false);
        cacc = __builtin_amdgcn_wmma_f32_16x16x32_f16(
            false, A1, false, Bf[nb][1], (short)0, cacc, false, false);
#pragma unroll
        for (int v = 0; v < 8; ++v) {
          const float h2 = fast_tanh(cacc[v] + b2v[nb]);
          acc[v] = fmaf(h2, w3v[nb], acc[v]);   // partial row-dot with W3
        }
      }

      // ---------- reduce over the 16 columns held by each half-wave ------
#pragma unroll
      for (int m = 1; m <= 8; m <<= 1) {
#pragma unroll
        for (int v = 0; v < 8; ++v)
          acc[v] += __shfl_xor(acc[v], m, 32);
      }

      // ---------- register-only redistribution: kcur = k[row lane16] -----
      // acc[v] holds k for row (v + 8*half); select acc[sel], then lanes
      // 8..23 take their value from the opposite half via shfl_xor(16).
      float tmp = acc[0];
#pragma unroll
      for (int v = 1; v < 8; ++v)
        tmp = (sel == v) ? acc[v] : tmp;
      const float sw   = __shfl_xor(tmp, 16, 32);
      const float kcur = (midlane ? sw : tmp) + b3s;

      kv[s] = kcur;
      if (s == 0)      ya = fmaf(0.5f, kcur, y);  // arg for k2
      else if (s == 1) ya = fmaf(0.5f, kcur, y);  // arg for k3
      else if (s == 2) ya = y + kcur;             // arg for k4
    }

    y = fmaf(kv[0] + 2.0f * kv[1] + 2.0f * kv[2] + kv[3], (1.0f / 6.0f), y);

    if (lane < 16) out[t * BATCH + row] = y;

    vin = vin_next;  // rotate prefetched x (wait lands here, fully hidden)
  }
}

extern "C" void kernel_launch(void* const* d_in, const int* in_sizes, int n_in,
                              void* d_out, int out_size, void* d_ws, size_t ws_size,
                              hipStream_t stream) {
  (void)in_sizes; (void)n_in; (void)d_ws; (void)ws_size; (void)out_size;
  const float* x  = (const float*)d_in[0];
  const float* W1 = (const float*)d_in[1];
  const float* b1 = (const float*)d_in[2];
  const float* W2 = (const float*)d_in[3];
  const float* b2 = (const float*)d_in[4];
  const float* W3 = (const float*)d_in[5];
  const float* b3 = (const float*)d_in[6];
  float* out = (float*)d_out;

  dim3 grid(BATCH / 16);   // 64 waves total == max useful parallelism
  dim3 block(32);          // one wave32 per workgroup
  rk4_scan_kernel<<<grid, block, 0, stream>>>(x, W1, b1, W2, b2, W3, b3, out);
}
